// CrazyNet_90039694393584
// MI455X (gfx1250) — matrix operand order repaired and checked by reading, not compiled, not run on hardware
//
#include <hip/hip_runtime.h>

// ---------------------------------------------------------------------------
// Problem constants
// ---------------------------------------------------------------------------
#define B_    4
#define N_    1023
#define S_    1024
#define CD_   3
#define FD_   8
#define DM_   512
#define H_    8
#define L_    4
#define DFF_  2048
#define BS_   (B_ * S_)        // 4096 rows

// GEMM epilogue flags
#define GEMM_RELU   1
#define GEMM_WF32   2
#define GEMM_WBF16  4
#define GEMM_WBF16T 8          // write bf16 transposed: ChT[col*M + row]

typedef __attribute__((ext_vector_type(16))) __bf16 v16bf;
typedef __attribute__((ext_vector_type(8)))  __bf16 v8bf;
typedef __attribute__((ext_vector_type(8)))  float  v8f;
typedef __attribute__((ext_vector_type(4)))  unsigned int u32x4;
typedef __attribute__((ext_vector_type(8)))  int    i32x8;
typedef __attribute__((ext_vector_type(4)))  int    i32x4;

// ---------------------------------------------------------------------------
// Tensor Data Mover: 2D tile (tile_w x tile_h elements) global -> LDS.
// D# group0: count=1 | lds_addr | global_addr(57b) | type=2
// D# group1: data_size | tensor_dim0/1 | tile_dim0/1 | tensor_dim0_stride
// Issued per-wave (EXEC ignored) -> call from wave 0 only.
// ---------------------------------------------------------------------------
__device__ __forceinline__ void tdm_load_2d(unsigned lds_off, const void* gptr,
                                            unsigned tile_w, unsigned tile_h,
                                            unsigned tensor_w, unsigned tensor_h,
                                            unsigned stride_elems, unsigned dsize_code)
{
    unsigned long long ga = (unsigned long long)(size_t)gptr;
    u32x4 g0;
    g0[0] = 1u;                                                  // count=1, user D#
    g0[1] = lds_off;                                             // lds_addr
    g0[2] = (unsigned)ga;                                        // global_addr[31:0]
    g0[3] = (unsigned)((ga >> 32) & 0x1FFFFFFu) | (2u << 30);    // addr[56:32] | type=2
    i32x8 g1;
    g1[0] = (int)(dsize_code << 16);                             // wg_mask=0, data_size
    g1[1] = (int)((tensor_w & 0xFFFFu) << 16);                   // barrier=0, dim0 lo
    g1[2] = (int)((tensor_w >> 16) | ((tensor_h & 0xFFFFu) << 16));
    g1[3] = (int)((tensor_h >> 16) | (tile_w << 16));            // dim1 hi | tile_dim0
    g1[4] = (int)(tile_h & 0xFFFFu);                             // tile_dim1, tile_dim2=0
    g1[5] = (int)stride_elems;                                   // dim0_stride lo32
    g1[6] = 0;
    g1[7] = 0;
    i32x4 z4 = {0, 0, 0, 0};
    i32x8 z8 = {0, 0, 0, 0, 0, 0, 0, 0};
    __builtin_amdgcn_tensor_load_to_lds(g0, g1, z4, z4, z8, 0);
}

__device__ __forceinline__ unsigned lds_offset(const void* p)
{
    // Generic LDS address = {SHARED_BASE, offset32}; low 32 bits are the LDS offset.
    return (unsigned)(size_t)p;
}

// A-fragment (16x32 bf16): lane half c0 = 0/16; per lane 16 contiguous
// K at [c0, c0+16).
__device__ __forceinline__ v16bf load_a_frag(const __bf16* row, int c0)
{
    v16bf a;
    v8bf lo = *(const v8bf*)(row + c0);
    v8bf hi = *(const v8bf*)(row + c0 + 8);
#pragma unroll
    for (int e = 0; e < 8; ++e) { a[e] = lo[e]; a[8 + e] = hi[e]; }
    return a;
}

// B-fragment (32x16 bf16) from an N-major (transposed) LDS tile: lane holds
// column N=lane%16 with K k0b..k0b+15 contiguous -> one 32-byte load.
__device__ __forceinline__ v16bf load_b_frag_t(const __bf16* colrow, int k0b)
{
    return *(const v16bf*)(colrow + k0b);
}

// ---------------------------------------------------------------------------
// Generic tiled bf16 GEMM: C[M,N] = A[M,K] * Bt[N,K]^T + bias, optional relu,
// writes f32 / bf16 / bf16-transposed. Tiles: 128x64, K-step 32, 8 waves of
// 32x32 each. A/B tiles loaded with the TDM; B is stored N-major so every
// WMMA fragment is a contiguous LDS load (no gathers).
// ---------------------------------------------------------------------------
__global__ __launch_bounds__(256) void gemm_bf16_kernel(
    const __bf16* __restrict__ A, const __bf16* __restrict__ Bt,
    const float* __restrict__ bias, float* __restrict__ Cf,
    __bf16* __restrict__ Ch, __bf16* __restrict__ ChT,
    int M, int N, int K, int flags)
{
    __shared__ alignas(128) __bf16 As[128 * 32];
    __shared__ alignas(128) __bf16 Bs[64 * 32];     // N-major: [n][k]

    const int tid = threadIdx.x, wave = tid >> 5, lane = tid & 31;
    const int wm = wave & 3, wn = wave >> 2;        // 4 x 2 wave grid
    const int m0 = blockIdx.y * 128, n0 = blockIdx.x * 64;
    const int nf  = lane & 15;
    const int c0a = (lane < 16) ? 0 : 16;
    const int k0b = (lane < 16) ? 0 : 16;

    const unsigned as_off = lds_offset(As);
    const unsigned bs_off = lds_offset(Bs);

    v8f acc[2][2] = {};

    for (int k0 = 0; k0 < K; k0 += 32) {
        __syncthreads();                            // prior compute done
        if (tid < 32) {                             // wave 0 drives the TDM
            tdm_load_2d(as_off, A  + (size_t)m0 * K + k0, 32, 128,
                        (unsigned)K, (unsigned)M, (unsigned)K, 1u);
            tdm_load_2d(bs_off, Bt + (size_t)n0 * K + k0, 32, 64,
                        (unsigned)K, (unsigned)N, (unsigned)K, 1u);
            __builtin_amdgcn_s_wait_tensorcnt(0);
        }
        __syncthreads();                            // tiles visible to all waves

#pragma unroll
        for (int i = 0; i < 2; ++i) {
            v16bf a = load_a_frag(As + (wm * 32 + i * 16 + nf) * 32, c0a);
#pragma unroll
            for (int j = 0; j < 2; ++j) {
                v16bf b = load_b_frag_t(Bs + (wn * 32 + j * 16 + nf) * 32, k0b);
                acc[i][j] = __builtin_amdgcn_wmma_f32_16x16x32_bf16(
                    false, a, false, b, (short)0, acc[i][j], false, false);
            }
        }
    }

    const int rb = (lane < 16) ? 0 : 8;
#pragma unroll
    for (int i = 0; i < 2; ++i)
#pragma unroll
        for (int j = 0; j < 2; ++j)
#pragma unroll
            for (int r = 0; r < 8; ++r) {
                int row = m0 + wm * 32 + i * 16 + rb + r;
                int col = n0 + wn * 32 + j * 16 + nf;
                float v = acc[i][j][r] + (bias ? bias[col] : 0.f);
                if (flags & GEMM_RELU) v = fmaxf(v, 0.f);
                if (flags & GEMM_WF32)   Cf[(size_t)row * N + col] = v;
                if (flags & GEMM_WBF16)  Ch[(size_t)row * N + col] = (__bf16)v;
                if (flags & GEMM_WBF16T) ChT[(size_t)col * M + row] = (__bf16)v;
            }
}

// ---------------------------------------------------------------------------
// Fused attention, one block per (b, head, 128-query tile); 8 waves, each wave
// owns 16 query rows. Online (flash) softmax with the distance-bias logits'
// row statistics living in the WMMA C-layout lane halves. V is consumed
// pre-transposed (vT[depth][b*S+s]) so the P@V B-fragment is contiguous.
// ---------------------------------------------------------------------------
__global__ __launch_bounds__(256) void attn_kernel(
    const __bf16* __restrict__ q, const __bf16* __restrict__ k,
    const __bf16* __restrict__ vT,
    const float* __restrict__ alpha, const float* __restrict__ beta,
    const float* __restrict__ xd, __bf16* __restrict__ o)
{
    __shared__ alignas(128) __bf16 Qs[128 * 64];
    __shared__ alignas(128) __bf16 Ks[64 * 64];      // [key][depth]
    __shared__ alignas(128) __bf16 Vs[64 * 64];      // [depth][key]  (from vT)
    __shared__ alignas(128) __bf16 Ps[8][16 * 64];   // per-wave P staging

    const int tid = threadIdx.x, wave = tid >> 5, lane = tid & 31;
    const int blk = blockIdx.x;
    const int qt = blk & 7, hh = (blk >> 3) & 7, b = blk >> 6;
    const int q0 = qt * 128;
    const int nf  = lane & 15;
    const int c0a = (lane < 16) ? 0 : 16;
    const int k0b = (lane < 16) ? 0 : 16;
    const int rb  = (lane < 16) ? 0 : 8;

    if (tid < 32) {
        tdm_load_2d(lds_offset(Qs), q + ((size_t)(b * S_ + q0)) * DM_ + hh * 64,
                    64, 128, DM_, BS_, DM_, 1u);
        __builtin_amdgcn_s_wait_tensorcnt(0);
    }

    float al[8], bt[8], mrow[8], lrow[8];
    v8f Oacc[4] = {};
    const size_t absbase = ((size_t)b * H_ + hh) * S_ + q0 + wave * 16 + rb;
#pragma unroll
    for (int r = 0; r < 8; ++r) {
        al[r] = alpha[absbase + r];
        bt[r] = beta[absbase + r];
        mrow[r] = -3.0e38f;
        lrow[r] = 0.f;
    }
    __syncthreads();

    const int qrow_g = q0 + wave * 16 + rb;

    for (int kt = 0; kt < 16; ++kt) {
        if (tid < 32) {
            tdm_load_2d(lds_offset(Ks), k + ((size_t)(b * S_ + kt * 64)) * DM_ + hh * 64,
                        64, 64, DM_, BS_, DM_, 1u);
            tdm_load_2d(lds_offset(Vs), vT + (size_t)(hh * 64) * BS_ + b * S_ + kt * 64,
                        64, 64, BS_, DM_, BS_, 1u);
            __builtin_amdgcn_s_wait_tensorcnt(0);
        }
        __syncthreads();

        // S = Q K^T  (per wave: 16 q-rows x 64 keys)
        v8f Sacc[4] = {};
#pragma unroll
        for (int kk = 0; kk < 64; kk += 32) {
            v16bf a = load_a_frag(Qs + (wave * 16 + nf) * 64 + kk, c0a);
#pragma unroll
            for (int nt = 0; nt < 4; ++nt) {
                v16bf bb = *(const v16bf*)(Ks + (nt * 16 + nf) * 64 + kk + k0b);
                Sacc[nt] = __builtin_amdgcn_wmma_f32_16x16x32_bf16(
                    false, a, false, bb, (short)0, Sacc[nt], false, false);
            }
        }

        // scale + alpha*xd + beta*xd^2, track new row max
        float mnew[8];
#pragma unroll
        for (int r = 0; r < 8; ++r) mnew[r] = mrow[r];
#pragma unroll
        for (int nt = 0; nt < 4; ++nt) {
            const int kc = kt * 64 + nt * 16 + nf;
#pragma unroll
            for (int r = 0; r < 8; ++r) {
                float xv = xd[((size_t)b * S_ + qrow_g + r) * S_ + kc];
                float vv = Sacc[nt][r] * 0.125f + al[r] * xv + bt[r] * (xv * xv);
                Sacc[nt][r] = vv;
                mnew[r] = fmaxf(mnew[r], vv);
            }
        }

        // online softmax (16-lane row groups match the WMMA C-layout)
#pragma unroll
        for (int r = 0; r < 8; ++r) {
            float m = mnew[r];
#pragma unroll
            for (int o2 = 1; o2 < 16; o2 <<= 1) m = fmaxf(m, __shfl_xor(m, o2, 16));
            float fs = __expf(mrow[r] - m);
            mrow[r] = m;
            float rs = 0.f;
#pragma unroll
            for (int nt = 0; nt < 4; ++nt) {
                float p = __expf(Sacc[nt][r] - m);
                Sacc[nt][r] = p;
                rs += p;
            }
#pragma unroll
            for (int o2 = 1; o2 < 16; o2 <<= 1) rs += __shfl_xor(rs, o2, 16);
            lrow[r] = lrow[r] * fs + rs;
#pragma unroll
            for (int dt = 0; dt < 4; ++dt) Oacc[dt][r] *= fs;
        }

        // stage P (C-layout -> LDS row-major) for the A-fragment of P@V
        __bf16* Pw = &Ps[wave][0];
#pragma unroll
        for (int nt = 0; nt < 4; ++nt)
#pragma unroll
            for (int r = 0; r < 8; ++r)
                Pw[(rb + r) * 64 + nt * 16 + nf] = (__bf16)Sacc[nt][r];

        // O += P V  (B-frag contiguous from transposed V tile)
#pragma unroll
        for (int kk = 0; kk < 64; kk += 32) {
            v16bf a = load_a_frag(Pw + nf * 64 + kk, c0a);
#pragma unroll
            for (int dt = 0; dt < 4; ++dt) {
                v16bf bb = *(const v16bf*)(Vs + (dt * 16 + nf) * 64 + kk + k0b);
                Oacc[dt] = __builtin_amdgcn_wmma_f32_16x16x32_bf16(
                    false, a, false, bb, (short)0, Oacc[dt], false, false);
            }
        }
        __syncthreads();            // done with Ks/Vs before next TDM overwrite
    }

#pragma unroll
    for (int r = 0; r < 8; ++r) {
        float inv = 1.f / lrow[r];
        const size_t rowoff = ((size_t)(b * S_) + qrow_g + r) * DM_ + hh * 64;
#pragma unroll
        for (int dt = 0; dt < 4; ++dt)
            o[rowoff + dt * 16 + nf] = (__bf16)(Oacc[dt][r] * inv);
    }
}

// ---------------------------------------------------------------------------
// Small helper kernels
// ---------------------------------------------------------------------------
// Convert f32 [L][K][N] weights to bf16 stored transposed per layer: [L][N][K]
__global__ void cvt_f32_bf16_t_kernel(const float* __restrict__ src,
                                      __bf16* __restrict__ dst, int K, int N)
{
    int idx = blockIdx.x * 256 + threadIdx.x;
    if (idx >= L_ * K * N) return;
    int n = idx % N;
    int kk = (idx / N) % K;
    int l = idx / (N * K);
    dst[(size_t)l * K * N + (size_t)n * K + kk] = (__bf16)src[idx];
}

__global__ void xd_kernel(const float* __restrict__ x,
                          const float* __restrict__ x_inputs,
                          float* __restrict__ xd)
{
    int idx = blockIdx.x * 256 + threadIdx.x;          // B*S*S = 4M
    int j = idx & (S_ - 1);
    int i = (idx >> 10) & (S_ - 1);
    int b = idx >> 20;
    float d2 = 0.f;
#pragma unroll
    for (int c = 0; c < CD_; ++c) {
        float xi = (i == 0) ? x[b * CD_ + c]
                            : x_inputs[((size_t)b * N_ + (i - 1)) * CD_ + c];
        float xj = (j == 0) ? x[b * CD_ + c]
                            : x_inputs[((size_t)b * N_ + (j - 1)) * CD_ + c];
        float d = xi - xj;
        d2 += d * d;
    }
    xd[idx] = sqrtf(d2);
}

__global__ void init_h_kernel(const float* __restrict__ x_token,
                              const float* __restrict__ inputs,
                              const float* __restrict__ W_in,
                              const float* __restrict__ b_in,
                              float* __restrict__ h, __bf16* __restrict__ hb)
{
    int idx = blockIdx.x * 256 + threadIdx.x;          // B*S*DM = 2M
    int d = idx & (DM_ - 1);
    int s = (idx >> 9) & (S_ - 1);
    int b = idx >> 19;
    float v;
    if (s == 0) {
        v = x_token[d];
    } else {
        const float* ip = inputs + ((size_t)b * N_ + (s - 1)) * FD_;
        float a = b_in[d];
#pragma unroll
        for (int kk = 0; kk < FD_; ++kk) a += ip[kk] * W_in[kk * DM_ + d];
        v = fmaxf(a, 0.f);
    }
    h[idx] = v;
    hb[idx] = (__bf16)v;
}

__global__ void alphabeta_kernel(const __bf16* __restrict__ q,
                                 const float* __restrict__ Wa, const float* __restrict__ ba,
                                 const float* __restrict__ Wb, const float* __restrict__ bb,
                                 float* __restrict__ alpha, float* __restrict__ beta)
{
    int idx = blockIdx.x * 256 + threadIdx.x;          // B*S*H = 32768
    int hh = idx & 7;
    int s  = (idx >> 3) & (S_ - 1);
    int b  = idx >> 13;
    const __bf16* qp = q + ((size_t)(b * S_ + s)) * DM_;
    float a = ba[hh], bt = bb[hh];
    for (int kk = 0; kk < DM_; ++kk) {
        float qq = (float)qp[kk];
        a  += qq * Wa[kk * H_ + hh];
        bt += qq * Wb[kk * H_ + hh];
    }
    alpha[((size_t)b * H_ + hh) * S_ + s] = a;
    beta [((size_t)b * H_ + hh) * S_ + s] = bt;
}

__global__ __launch_bounds__(256) void add_ln_kernel(
    const float* __restrict__ hin, const float* __restrict__ res,
    const float* __restrict__ g, const float* __restrict__ be,
    float* __restrict__ hout, __bf16* __restrict__ hb)
{
    int bs = blockIdx.x, tid = threadIdx.x, lane = tid & 31, wv = tid >> 5;
    const size_t base = (size_t)bs * DM_;
    float v0 = hin[base + tid]       + res[base + tid];
    float v1 = hin[base + 256 + tid] + res[base + 256 + tid];
    __shared__ float p1[8], p2[8];
    float s = v0 + v1;
#pragma unroll
    for (int o2 = 1; o2 < 32; o2 <<= 1) s += __shfl_xor(s, o2, 32);
    if (lane == 0) p1[wv] = s;
    __syncthreads();
    float mean = 0.f;
#pragma unroll
    for (int i = 0; i < 8; ++i) mean += p1[i];
    mean *= (1.f / DM_);
    float d0 = v0 - mean, d1 = v1 - mean;
    float qv = d0 * d0 + d1 * d1;
#pragma unroll
    for (int o2 = 1; o2 < 32; o2 <<= 1) qv += __shfl_xor(qv, o2, 32);
    if (lane == 0) p2[wv] = qv;
    __syncthreads();
    float var = 0.f;
#pragma unroll
    for (int i = 0; i < 8; ++i) var += p2[i];
    var *= (1.f / DM_);
    float inv = rsqrtf(var + 1e-6f);
    float y0 = d0 * inv * g[tid]       + be[tid];
    float y1 = d1 * inv * g[256 + tid] + be[256 + tid];
    hout[base + tid] = y0;       hout[base + 256 + tid] = y1;
    hb[base + tid] = (__bf16)y0; hb[base + 256 + tid] = (__bf16)y1;
}

__global__ __launch_bounds__(256) void head_kernel(
    const float* __restrict__ h,
    const float* __restrict__ Wp1, const float* __restrict__ bp1,
    const float* __restrict__ Wp2, const float* __restrict__ bp2,
    const float* __restrict__ Wf,  const float* __restrict__ bf_,
    float* __restrict__ out)
{
    __shared__ float val[B_ * S_];
    __shared__ float t1[B_ * 512];
    __shared__ float t2[B_ * 256];
    int tid = threadIdx.x;
    for (int i = tid; i < B_ * S_; i += 256) val[i] = h[(size_t)i * DM_];  // h[...,0]
    __syncthreads();
    for (int i = tid; i < B_ * 512; i += 256) {
        int b = i >> 9, c = i & 511;
        float a = bp1[c];
        for (int kk = 0; kk < S_; ++kk) a += val[b * S_ + kk] * Wp1[kk * 512 + c];
        t1[i] = fmaxf(a, 0.f);
    }
    __syncthreads();
    for (int i = tid; i < B_ * 256; i += 256) {
        int b = i >> 8, c = i & 255;
        float a = bp2[c];
        for (int kk = 0; kk < 512; ++kk) a += t1[b * 512 + kk] * Wp2[kk * 256 + c];
        t2[i] = fmaxf(a, 0.f);
    }
    __syncthreads();
    if (tid < B_) {
        float a = bf_[0];
        for (int kk = 0; kk < 256; ++kk) a += t2[tid * 256 + kk] * Wf[kk];
        out[tid] = a;
    }
}

// ---------------------------------------------------------------------------
// Host driver
// ---------------------------------------------------------------------------
extern "C" void kernel_launch(void* const* d_in, const int* in_sizes, int n_in,
                              void* d_out, int out_size, void* d_ws, size_t ws_size,
                              hipStream_t stream)
{
    const float* x        = (const float*)d_in[0];
    const float* x_inputs = (const float*)d_in[1];
    const float* inputs   = (const float*)d_in[2];
    const float* W_in     = (const float*)d_in[3];
    const float* b_in     = (const float*)d_in[4];
    const float* x_token  = (const float*)d_in[5];
    const float* Wq = (const float*)d_in[6];   const float* bq = (const float*)d_in[7];
    const float* Wk = (const float*)d_in[8];   const float* bk = (const float*)d_in[9];
    const float* Wv = (const float*)d_in[10];  const float* bv = (const float*)d_in[11];
    const float* Wa = (const float*)d_in[12];  const float* ba = (const float*)d_in[13];
    const float* Wb = (const float*)d_in[14];  const float* bb = (const float*)d_in[15];
    const float* Wo = (const float*)d_in[16];  const float* bo = (const float*)d_in[17];
    const float* W1 = (const float*)d_in[18];  const float* b1 = (const float*)d_in[19];
    const float* W2 = (const float*)d_in[20];  const float* b2 = (const float*)d_in[21];
    const float* g1 = (const float*)d_in[22];  const float* be1 = (const float*)d_in[23];
    const float* g2 = (const float*)d_in[24];  const float* be2 = (const float*)d_in[25];
    const float* Wp1 = (const float*)d_in[26]; const float* bp1 = (const float*)d_in[27];
    const float* Wp2 = (const float*)d_in[28]; const float* bp2 = (const float*)d_in[29];
    const float* Wf  = (const float*)d_in[30]; const float* bf_ = (const float*)d_in[31];
    float* out = (float*)d_out;
    (void)in_sizes; (void)n_in; (void)out_size; (void)ws_size;

    char* base = (char*)d_ws;
    size_t off = 0;
    auto carve = [&](size_t bytes) -> void* {
        void* p = base + off;
        off = (off + bytes + 255) & ~(size_t)255;
        return p;
    };
    float*  h_f   = (float*)carve((size_t)BS_ * DM_ * 4);
    float*  tmp_f = (float*)carve((size_t)BS_ * DM_ * 4);
    float*  xd    = (float*)carve((size_t)B_ * S_ * S_ * 4);
    float*  alpha = (float*)carve((size_t)B_ * H_ * S_ * 4);
    float*  beta  = (float*)carve((size_t)B_ * H_ * S_ * 4);
    __bf16* h_b   = (__bf16*)carve((size_t)BS_ * DM_ * 2);
    __bf16* q_b   = (__bf16*)carve((size_t)BS_ * DM_ * 2);
    __bf16* k_b   = (__bf16*)carve((size_t)BS_ * DM_ * 2);
    __bf16* vT_b  = (__bf16*)carve((size_t)BS_ * DM_ * 2);   // [DM][BS]
    __bf16* o_b   = (__bf16*)carve((size_t)BS_ * DM_ * 2);
    __bf16* ff_b  = (__bf16*)carve((size_t)BS_ * DFF_ * 2);
    __bf16* wqT   = (__bf16*)carve((size_t)L_ * DM_ * DM_ * 2);
    __bf16* wkT   = (__bf16*)carve((size_t)L_ * DM_ * DM_ * 2);
    __bf16* wvT   = (__bf16*)carve((size_t)L_ * DM_ * DM_ * 2);
    __bf16* woT   = (__bf16*)carve((size_t)L_ * DM_ * DM_ * 2);
    __bf16* w1T   = (__bf16*)carve((size_t)L_ * DM_ * DFF_ * 2);
    __bf16* w2T   = (__bf16*)carve((size_t)L_ * DFF_ * DM_ * 2);

    // ---- weight conversion to bf16, stored transposed (N-major) ----
    const int nW = L_ * DM_ * DM_;
    const int nF = L_ * DM_ * DFF_;
    cvt_f32_bf16_t_kernel<<<(nW + 255) / 256, 256, 0, stream>>>(Wq, wqT, DM_, DM_);
    cvt_f32_bf16_t_kernel<<<(nW + 255) / 256, 256, 0, stream>>>(Wk, wkT, DM_, DM_);
    cvt_f32_bf16_t_kernel<<<(nW + 255) / 256, 256, 0, stream>>>(Wv, wvT, DM_, DM_);
    cvt_f32_bf16_t_kernel<<<(nW + 255) / 256, 256, 0, stream>>>(Wo, woT, DM_, DM_);
    cvt_f32_bf16_t_kernel<<<(nF + 255) / 256, 256, 0, stream>>>(W1, w1T, DM_, DFF_);
    cvt_f32_bf16_t_kernel<<<(nF + 255) / 256, 256, 0, stream>>>(W2, w2T, DFF_, DM_);

    // ---- precompute distance matrix and initial hidden state ----
    xd_kernel<<<(B_ * S_ * S_) / 256, 256, 0, stream>>>(x, x_inputs, xd);
    init_h_kernel<<<(BS_ * DM_) / 256, 256, 0, stream>>>(x_token, inputs, W_in, b_in, h_f, h_b);

    const dim3 gDM(DM_ / 64, BS_ / 128);   // (8, 32)
    const dim3 gFF(DFF_ / 64, BS_ / 128);  // (32, 32)

    for (int i = 0; i < L_; ++i) {
        const size_t wOff = (size_t)i * DM_ * DM_;
        const size_t fOff = (size_t)i * DM_ * DFF_;
        gemm_bf16_kernel<<<gDM, 256, 0, stream>>>(h_b, wqT + wOff, bq + i * DM_,
                                                  nullptr, q_b, nullptr,
                                                  BS_, DM_, DM_, GEMM_WBF16);
        gemm_bf16_kernel<<<gDM, 256, 0, stream>>>(h_b, wkT + wOff, bk + i * DM_,
                                                  nullptr, k_b, nullptr,
                                                  BS_, DM_, DM_, GEMM_WBF16);
        gemm_bf16_kernel<<<gDM, 256, 0, stream>>>(h_b, wvT + wOff, bv + i * DM_,
                                                  nullptr, nullptr, vT_b,
                                                  BS_, DM_, DM_, GEMM_WBF16T);
        alphabeta_kernel<<<(B_ * S_ * H_) / 256, 256, 0, stream>>>(
            q_b, Wa + (size_t)i * DM_ * H_, ba + i * H_,
            Wb + (size_t)i * DM_ * H_, bb + i * H_, alpha, beta);
        attn_kernel<<<B_ * H_ * (S_ / 128), 256, 0, stream>>>(q_b, k_b, vT_b,
                                                              alpha, beta, xd, o_b);
        gemm_bf16_kernel<<<gDM, 256, 0, stream>>>(o_b, woT + wOff, bo + i * DM_,
                                                  tmp_f, nullptr, nullptr,
                                                  BS_, DM_, DM_, GEMM_WF32);
        add_ln_kernel<<<BS_, 256, 0, stream>>>(h_f, tmp_f, g1 + i * DM_, be1 + i * DM_, h_f, h_b);
        gemm_bf16_kernel<<<gFF, 256, 0, stream>>>(h_b, w1T + fOff, b1 + i * DFF_,
                                                  nullptr, ff_b, nullptr,
                                                  BS_, DFF_, DM_, GEMM_RELU | GEMM_WBF16);
        gemm_bf16_kernel<<<gDM, 256, 0, stream>>>(ff_b, w2T + (size_t)i * DFF_ * DM_,
                                                  b2 + i * DM_, tmp_f, nullptr, nullptr,
                                                  BS_, DM_, DFF_, GEMM_WF32);
        add_ln_kernel<<<BS_, 256, 0, stream>>>(h_f, tmp_f, g2 + i * DM_, be2 + i * DM_, h_f, h_b);
    }

    head_kernel<<<1, 256, 0, stream>>>(h_f, Wp1, bp1, Wp2, bp2, Wf, bf_, out);
}
